// FusedAttentionWithALiBi_38671885534082
// MI455X (gfx1250) — matrix operand order repaired and checked by reading, not compiled, not run on hardware
//
#include <hip/hip_runtime.h>
#include <hip/hip_bf16.h>

// ---------------------------------------------------------------------------
// CDNA5 (gfx1250) fused attention w/ ALiBi.
// Pipeline: cvt(f32->bf16) -> WMMA GEMM (QKV, bias fused) -> flash-attention
// (WMMA bf16, online softmax, ALiBi+causal) -> WMMA GEMM (out proj, f32 out).
// All matrix math via v_wmma_f32_16x16x32_bf16 (wave32).
// Double-buffered LDS staging (1 barrier/tile); async global->LDS (b128).
// ---------------------------------------------------------------------------

typedef unsigned short u16;
typedef __attribute__((ext_vector_type(16))) __bf16        v16bf;
typedef __attribute__((ext_vector_type(8)))  float         v8f;
typedef __attribute__((ext_vector_type(4)))  unsigned int  v4u;

union Frag16 {          // one WMMA 16-bit operand fragment (8 VGPRs / lane)
    v16bf v;
    v4u   q[2];
    u16   h[16];
    unsigned int u[8];
};

__device__ __forceinline__ u16 f2bf(float f) {   // round-to-nearest-even
    unsigned int u = __float_as_uint(f);
    unsigned int r = u + 0x7FFFu + ((u >> 16) & 1u);
    return (u16)(r >> 16);
}

__device__ __forceinline__ v8f v8f_zero() {
    v8f z = {0.f, 0.f, 0.f, 0.f, 0.f, 0.f, 0.f, 0.f};
    return z;
}

// ---- async global->LDS path (GLOBAL_LOAD_ASYNC_TO_LDS_B128) ---------------
// Probe-verified: param0 is `int __vector(4) addrspace(1)*` (global src).
#if defined(__gfx1250__) && __has_builtin(__builtin_amdgcn_global_load_async_to_lds_b128)
#define HAVE_ASYNC_LDS 1
typedef int v4i_vs __attribute__((vector_size(4 * sizeof(int))));
typedef __attribute__((address_space(1))) v4i_vs* as1_v4i;
typedef __attribute__((address_space(3))) v4i_vs* as3_v4i;
#endif

// 16-byte global -> LDS copy (async when available, else load+store)
__device__ __forceinline__ void cp128(const u16* __restrict__ g, u16* l) {
#if HAVE_ASYNC_LDS
    __builtin_amdgcn_global_load_async_to_lds_b128(
        (as1_v4i)(u16*)g, (as3_v4i)l, 0, 0);
#else
    *(v4u*)l = *(const v4u*)g;
#endif
}

__device__ __forceinline__ void async_wait() {
#if HAVE_ASYNC_LDS
#if __has_builtin(__builtin_amdgcn_s_wait_asynccnt)
    __builtin_amdgcn_s_wait_asynccnt(0);
#else
    asm volatile("s_wait_asynccnt 0" ::: "memory");
#endif
#endif
}

// ---------------------------------------------------------------------------
// Kernel 1: f32 -> bf16 conversion (8 elements / thread, b128 in / b128 out)
// ---------------------------------------------------------------------------
__global__ __launch_bounds__(256)
void cvt_f32_bf16(const float* __restrict__ src, u16* __restrict__ dst, int n) {
    int i = (blockIdx.x * 256 + threadIdx.x) * 8;
    if (i >= n) return;
    float4 a = *(const float4*)(src + i);
    float4 b = *(const float4*)(src + i + 4);
    v4u pk;
    pk[0] = (unsigned)f2bf(a.x) | ((unsigned)f2bf(a.y) << 16);
    pk[1] = (unsigned)f2bf(a.z) | ((unsigned)f2bf(a.w) << 16);
    pk[2] = (unsigned)f2bf(b.x) | ((unsigned)f2bf(b.y) << 16);
    pk[3] = (unsigned)f2bf(b.z) | ((unsigned)f2bf(b.w) << 16);
    *(v4u*)(dst + i) = pk;
}

// ---------------------------------------------------------------------------
// Kernel 2/4: C[M,N] = A[M,K] @ Bw[N,K]^T + bias  (bf16 in, WMMA f32 acc)
// 128x128 block tile, 8 waves each 32(M)x64(N) = 8 accumulators.
// Double-buffered 128x32 LDS tiles (pad->40 halves: conflict-free frag reads).
// ---------------------------------------------------------------------------
__device__ __forceinline__ void store_val(float* p, float v) { *p = v; }
__device__ __forceinline__ void store_val(u16*   p, float v) { *p = f2bf(v); }

template <typename OutT>
__global__ __launch_bounds__(256)
void gemm_bf16_wmma(const u16* __restrict__ A,   // [M,K] bf16 row-major
                    const u16* __restrict__ Bw,  // [N,K] bf16 row-major
                    const float* __restrict__ bias, // [N]
                    OutT* __restrict__ C,        // [M,N]
                    int M, int N, int K) {
    __shared__ u16 As[2][128][40];
    __shared__ u16 Bs[2][128][40];

    const int tid  = threadIdx.x;
    const int lane = tid & 31;
    const int wave = tid >> 5;
    const int half = lane >> 4;
    const int l16  = lane & 15;
    const int m0 = blockIdx.y * 128;
    const int n0 = blockIdx.x * 128;
    const int wm = (wave >> 1) * 32;   // 4 waves along M
    const int wn = (wave & 1) * 64;    // 2 waves along N

    v8f acc[2][4];
    for (int i = 0; i < 2; ++i)
        for (int j = 0; j < 4; ++j) acc[i][j] = v8f_zero();

    auto stage = [&](int buf, int k0) {
        for (int p = 0; p < 2; ++p) {
            int idx = tid + p * 256;        // 0..511
            int row = idx >> 2;
            int kq  = (idx & 3) * 8;
            cp128(A  + (size_t)(m0 + row) * K + k0 + kq, &As[buf][row][kq]);
            cp128(Bw + (size_t)(n0 + row) * K + k0 + kq, &Bs[buf][row][kq]);
        }
    };

    stage(0, 0);
    for (int k0 = 0; k0 < K; k0 += 32) {
        const int cur = (k0 >> 5) & 1;
        async_wait();            // publish buf[cur] (async path)
        __syncthreads();         // all waves see buf[cur]; prev reads retired
        if (k0 + 32 < K) {
            stage(cur ^ 1, k0 + 32);   // overlap next tile with this compute
            __builtin_prefetch(A  + (size_t)(m0 + (tid >> 1)) * K + k0 + 64, 0, 1);
            __builtin_prefetch(Bw + (size_t)(n0 + (tid >> 1)) * K + k0 + 64, 0, 1);
        }

        Frag16 a[2], b[4];
        for (int i = 0; i < 2; ++i) {     // A frag: elems 0-7 @ half*8, 8-15 @ 16+half*8
            const u16* p = &As[cur][wm + i * 16 + l16][half * 8];
            a[i].q[0] = *(const v4u*)(p);
            a[i].q[1] = *(const v4u*)(p + 16);
        }
        for (int j = 0; j < 4; ++j) {     // B frag: elems 0-7 @ half*8, 8-15 @ 16+half*8
            const u16* p = &Bs[cur][wn + j * 16 + l16][half * 8];
            b[j].q[0] = *(const v4u*)(p);
            b[j].q[1] = *(const v4u*)(p + 16);
        }
        for (int i = 0; i < 2; ++i)
            for (int j = 0; j < 4; ++j)
                acc[i][j] = __builtin_amdgcn_wmma_f32_16x16x32_bf16(
                    false, a[i].v, false, b[j].v, (short)0, acc[i][j], false, false);
    }

    for (int i = 0; i < 2; ++i)
        for (int j = 0; j < 4; ++j) {
            int colg = n0 + wn + j * 16 + l16;
            float bv = bias[colg];
            for (int r = 0; r < 8; ++r) {
                int rowg = m0 + wm + i * 16 + r + half * 8;  // C layout: lane half splits M
                store_val(C + (size_t)rowg * N + colg, acc[i][j][r] + bv);
            }
        }
}

// ---------------------------------------------------------------------------
// Kernel 3: flash attention, one (b, h, 128-query tile) per workgroup.
// Scores computed transposed (M=keys, N=queries) so softmax stats live at
// lane==query and the score C-layout IS the A-fragment layout of P for P@V.
// K/V tiles double-buffered; V transposed at staging with packed u32 stores.
// ---------------------------------------------------------------------------
__global__ __launch_bounds__(256)
void flash_attn_wmma(const u16* __restrict__ qkv,   // [B*S, 6144]: [Q|K|V] x [H,128]
                     const float* __restrict__ slopes,
                     u16* __restrict__ attn_out,    // [B*S, 2048] bf16
                     int Slen) {
    __shared__ u16 Qs[128][136];      // query tile  [q][hd]
    __shared__ u16 Ks[2][32][136];    // key tile    [k][hd]
    __shared__ u16 Vt[2][128][40];    // value tile transposed [hd][k]

    const int tid  = threadIdx.x;
    const int lane = tid & 31;
    const int wave = tid >> 5;
    const int half = lane >> 4;
    const int l16  = lane & 15;
    const int qt = blockIdx.x, h = blockIdx.y, b = blockIdx.z;
    const int qbase = qt * 128;
    const float slope = slopes[h];
    const float scale = 0.08838834764831845f;    // 1/sqrt(128)

    const size_t rs = 6144;
    const u16* Qg = qkv + (size_t)b * Slen * rs + (size_t)h * 128;
    const u16* Kg = Qg + 2048;
    const u16* Vg = Qg + 4096;

    auto stage_kv = [&](int buf, int kbase) {
        // K: 32x128 tile, 512 b128 chunks, 2 per thread (async-capable)
        for (int c = 0; c < 2; ++c) {
            int ch  = tid * 2 + c;
            int row = ch >> 4;
            int col = (ch & 15) * 8;
            cp128(Kg + (size_t)(kbase + row) * rs + col, &Ks[buf][row][col]);
        }
        // V transposed: pack two adjacent key rows per u32 store
        int pr  = tid >> 4;            // key-pair index 0..15 -> rows 2pr,2pr+1
        int col = (tid & 15) * 8;      // hd base
        union { v4u q; u16 h[8]; } a, c;
        a.q = *(const v4u*)(Vg + (size_t)(kbase + 2 * pr)     * rs + col);
        c.q = *(const v4u*)(Vg + (size_t)(kbase + 2 * pr + 1) * rs + col);
        for (int i = 0; i < 8; ++i) {
            unsigned pk = (unsigned)a.h[i] | ((unsigned)c.h[i] << 16);
            *(unsigned*)(&Vt[buf][col + i][2 * pr]) = pk;
        }
    };

    // stage the 128x128 Q tile (async-capable) and the first K/V block
    {
        int row = tid >> 1, hh = (tid & 1) * 64;
        const u16* src = Qg + (size_t)(qbase + row) * rs + hh;
        for (int i = 0; i < 8; ++i)
            cp128(src + i * 8, &Qs[row][hh + i * 8]);
    }
    stage_kv(0, 0);
    async_wait();
    __syncthreads();

    // preload this wave's Q as B-fragments (reused every key block)
    Frag16 qf[4];
    for (int ks = 0; ks < 4; ++ks) {
        const u16* p = &Qs[wave * 16 + l16][ks * 32 + half * 16];
        qf[ks].q[0] = *(const v4u*)(p);
        qf[ks].q[1] = *(const v4u*)(p + 8);
    }

    v8f oacc[8];
    for (int nt = 0; nt < 8; ++nt) oacc[nt] = v8f_zero();
    float m_run = -3.0e38f, l_run = 0.0f;
    const int qg = qbase + wave * 16 + l16;        // this lane's query (score tiles)
    const int qwave_max = qbase + wave * 16 + 15;
    const int nkb = qt * 4 + 4;                    // causal: 32-key blocks needed

    for (int kb = 0; kb < nkb; ++kb) {
        const int kbase = kb * 32;
        const int cur = kb & 1;
        if (kb) { async_wait(); __syncthreads(); }   // publish buf[cur]
        if (kb + 1 < nkb) stage_kv(cur ^ 1, kbase + 32);

        if (kbase <= qwave_max) {       // no barriers inside; safe divergence
            float sc[2][8];
            for (int t2 = 0; t2 < 2; ++t2) {      // two 16-key score tiles
                v8f s = v8f_zero();
                for (int ks = 0; ks < 4; ++ks) {  // HD=128 in 4 k-steps
                    Frag16 ka;
                    const u16* p = &Ks[cur][t2 * 16 + l16][ks * 32 + half * 16];
                    ka.q[0] = *(const v4u*)(p);
                    ka.q[1] = *(const v4u*)(p + 8);
                    s = __builtin_amdgcn_wmma_f32_16x16x32_bf16(
                        false, ka.v, false, qf[ks].v, (short)0, s, false, false);
                }
                for (int r = 0; r < 8; ++r) {     // ALiBi + causal per element
                    int kg = kbase + t2 * 16 + r + half * 8;
                    float sv = s[r] * scale + slope * (float)(kg - qg);
                    sc[t2][r] = (kg <= qg) ? sv : -1.0e9f;
                }
            }
            // online softmax: stats per lane (lane==query), one half-swap shuffle
            float mb = sc[0][0];
            for (int r = 1; r < 8; ++r) mb = fmaxf(mb, sc[0][r]);
            for (int r = 0; r < 8; ++r) mb = fmaxf(mb, sc[1][r]);
            mb = fmaxf(mb, __shfl_xor(mb, 16, 32));
            float m_new = fmaxf(m_run, mb);
            float corr  = __expf(m_run - m_new);
            m_run = m_new;

            Frag16 pa;                             // P A-fragment: direct from score regs
            float bsum = 0.f;
            for (int r = 0; r < 8; ++r) {
                float e0 = __expf(sc[0][r] - m_new);
                float e1 = __expf(sc[1][r] - m_new);
                bsum += e0 + e1;
                pa.h[r]     = f2bf(e0);
                pa.h[r + 8] = f2bf(e1);
            }
            bsum += __shfl_xor(bsum, 16, 32);
            l_run = l_run * corr + bsum;

            // rescale output accumulators (query lives on VGPR axis there)
            for (int r = 0; r < 8; ++r) {
                float cr = __shfl(corr, r + ((lane & 16) >> 1), 32);
                for (int nt = 0; nt < 8; ++nt) oacc[nt][r] = oacc[nt][r] * cr;
            }
            // out += P @ V  (V B-frags: elems 0-7 @ half*8, 8-15 @ 16+half*8 from transposed LDS tile)
            for (int nt = 0; nt < 8; ++nt) {
                Frag16 vb;
                const u16* p = &Vt[cur][nt * 16 + l16][half * 8];
                vb.q[0] = *(const v4u*)(p);
                vb.q[1] = *(const v4u*)(p + 16);
                oacc[nt] = __builtin_amdgcn_wmma_f32_16x16x32_bf16(
                    false, pa.v, false, vb.v, (short)0, oacc[nt], false, false);
            }
        }
        if (kb + 1 < nkb) __syncthreads();   // writers of buf[cur^1] done before reuse
    }

    // normalize and store bf16 attention output
    float inv = 1.0f / l_run;
    u16* orow = attn_out + ((size_t)b * Slen + qbase + wave * 16) * 2048 + h * 128;
    for (int r = 0; r < 8; ++r) {
        float ir = __shfl(inv, r + ((lane & 16) >> 1), 32);
        int qrel = r + half * 8;
        for (int nt = 0; nt < 8; ++nt)
            orow[(size_t)qrel * 2048 + nt * 16 + l16] = f2bf(oacc[nt][r] * ir);
    }
}

// ---------------------------------------------------------------------------
// Host launcher
// ---------------------------------------------------------------------------
extern "C" void kernel_launch(void* const* d_in, const int* in_sizes, int n_in,
                              void* d_out, int out_size, void* d_ws, size_t ws_size,
                              hipStream_t stream) {
    const float* hs     = (const float*)d_in[0];  // [B,S,D]
    const float* qkv_w  = (const float*)d_in[1];  // [3D,D]
    const float* qkv_b  = (const float*)d_in[2];  // [3D]
    const float* out_w  = (const float*)d_in[3];  // [D,D]
    const float* out_b  = (const float*)d_in[4];  // [D]
    const float* slopes = (const float*)d_in[5];  // [H]
    float* out = (float*)d_out;

    const int B = 2, S = 2048, D = 2048, H = 16;
    const int MS = B * S;   // 4096 rows

    char* ws = (char*)d_ws;
    u16* hs_bf   = (u16*)ws; ws += (size_t)MS * D * 2;       // 16 MB
    u16* qkvw_bf = (u16*)ws; ws += (size_t)3 * D * D * 2;    // 24 MB
    u16* outw_bf = (u16*)ws; ws += (size_t)D * D * 2;        //  8 MB
    u16* qkv_bf  = (u16*)ws; ws += (size_t)MS * 3 * D * 2;   // 48 MB
    u16* attn_bf = (u16*)ws;                                 // 16 MB

    cvt_f32_bf16<<<(MS * D) / 2048, 256, 0, stream>>>(hs, hs_bf, MS * D);
    cvt_f32_bf16<<<(3 * D * D) / 2048, 256, 0, stream>>>(qkv_w, qkvw_bf, 3 * D * D);
    cvt_f32_bf16<<<(D * D) / 2048, 256, 0, stream>>>(out_w, outw_bf, D * D);

    gemm_bf16_wmma<u16><<<dim3(3 * D / 128, MS / 128), 256, 0, stream>>>(
        hs_bf, qkvw_bf, qkv_b, qkv_bf, MS, 3 * D, D);

    flash_attn_wmma<<<dim3(S / 128, H, B), 256, 0, stream>>>(
        qkv_bf, slopes, attn_bf, S);

    gemm_bf16_wmma<float><<<dim3(D / 128, MS / 128), 256, 0, stream>>>(
        attn_bf, outw_bf, out_b, out, MS, D, D);
}
